// Model_39676907885610
// MI455X (gfx1250) — matrix operand order repaired and checked by reading, not compiled, not run on hardware
//
#include <hip/hip_runtime.h>

typedef _Float16 v16h __attribute__((ext_vector_type(16)));
typedef _Float16 v8h  __attribute__((ext_vector_type(8)));
typedef float    v8f  __attribute__((ext_vector_type(8)));

#define WMMA_F32_F16(a, b, c) \
  __builtin_amdgcn_wmma_f32_16x16x32_f16(false, (a), false, (b), (short)0, (c), false, false)

// ---------------------------------------------------------------------------
// Fragment loaders per CDNA5 ISA 7.12.2 (wave32):
//  A (16x32 f16): lane = {row r = lane&15, hi = lane>>4}; per lane two 8-half
//    chunks at K = k0 + hi*8 and K = k0 + 16 + hi*8   (row-major source).
//  B (32x16 f16): lane = {col n = lane&15, hi = lane>>4}; per lane two 8-half
//    chunks at K = k0 + hi*8 and K = k0 + 16 + hi*8  (source laid out so
//    column n's K-values are contiguous: base + n*ld + k).
//  C/D (16x16 f32): vgpr i, lane l -> element (row = i + (l>>4)*8, col = l&15).
// ---------------------------------------------------------------------------
static __device__ __forceinline__ v16h load_a_frag(const _Float16* base, int ld,
                                                   int k0, int lane) {
  const int r = lane & 15, hi = (lane >> 4) & 1;
  const _Float16* p0 = base + r * ld + k0 + hi * 8;
  v16h f;
  ((v8h*)&f)[0] = *(const v8h*)p0;
  ((v8h*)&f)[1] = *(const v8h*)(p0 + 16);
  return f;
}

static __device__ __forceinline__ v16h load_b_frag(const _Float16* base, int ld,
                                                   int k0, int lane) {
  const int n = lane & 15, hi = (lane >> 4) & 1;
  const _Float16* p = base + n * ld + k0 + hi * 8;
  v16h f;
  ((v8h*)&f)[0] = *(const v8h*)p;
  ((v8h*)&f)[1] = *(const v8h*)(p + 16);
  return f;
}

// ---------------------------------------------------------------------------
// Stage 0: fp32 -> fp16 conversion (grid-stride)
// ---------------------------------------------------------------------------
__global__ void cvt_f32_f16(const float* __restrict__ src,
                            _Float16* __restrict__ dst, int n) {
  int i = blockIdx.x * blockDim.x + threadIdx.x;
  const int stride = gridDim.x * blockDim.x;
  for (; i < n; i += stride) dst[i] = (_Float16)src[i];
}

// ---------------------------------------------------------------------------
// Stage 1: fused QKV projection.  y = relu(x @ W^T + b) for W in {Wq,Wk,Wv}.
// Treated as one GEMM with N = 3*512 = 1536 (concatenated along N).
// Wave tile: 32 (M) x 64 (N): 2 A-frags + 4 B-frags -> 8 WMMAs per k-step.
//   x:  [8192, 512] f16 row-major (A-fragments contiguous)
//   W:  [512, 512] f16 row-major  (B column e == W row e, contiguous)
//   Q,K stored row-major f16 [8192,512]; V stored transposed Vt[b][d][s].
// tiles = (8192/32) * (1536/64) = 256*24 = 6144 -> grid 768 blocks, 8 waves.
// ---------------------------------------------------------------------------
__global__ void __launch_bounds__(256) qkv_gemm(
    const _Float16* __restrict__ xh,
    const _Float16* __restrict__ Wqh, const _Float16* __restrict__ Wkh,
    const _Float16* __restrict__ Wvh,
    const float* __restrict__ bq, const float* __restrict__ bk,
    const float* __restrict__ bv,
    _Float16* __restrict__ Qh, _Float16* __restrict__ Kh,
    _Float16* __restrict__ Vth) {
  const int tid  = threadIdx.x;
  const int wave = tid >> 5, lane = tid & 31;
  const int hi   = (lane >> 4) & 1, nlan = lane & 15;

  const int tile = blockIdx.x * 8 + wave;  // 0..6143
  const int tn = tile % 24, tm = tile / 24;
  const int m0  = tm * 32;
  const int gn0 = tn * 64;                 // global N in [0,1536)
  const int which = gn0 >> 9;              // 0=Q, 1=K, 2=V
  const int n0  = gn0 & 511;

  const _Float16* Wh = (which == 0) ? Wqh : (which == 1) ? Wkh : Wvh;
  const float* bias  = (which == 0) ? bq  : (which == 1) ? bk  : bv;

  const _Float16* A0 = xh + (size_t)m0 * 512;
  const _Float16* A1 = A0 + (size_t)16 * 512;
  const _Float16* B0 = Wh + (size_t)n0 * 512;

  v8f zero = {};
  v8f acc[2][4];
#pragma unroll
  for (int m = 0; m < 2; m++)
#pragma unroll
    for (int nt = 0; nt < 4; nt++) acc[m][nt] = zero;

#pragma unroll 4
  for (int k0 = 0; k0 < 512; k0 += 32) {
    v16h a0 = load_a_frag(A0, 512, k0, lane);
    v16h a1 = load_a_frag(A1, 512, k0, lane);
#pragma unroll
    for (int nt = 0; nt < 4; nt++) {
      v16h bf = load_b_frag(B0 + (size_t)nt * 16 * 512, 512, k0, lane);
      acc[0][nt] = WMMA_F32_F16(a0, bf, acc[0][nt]);
      acc[1][nt] = WMMA_F32_F16(a1, bf, acc[1][nt]);
    }
  }

#pragma unroll
  for (int nt = 0; nt < 4; nt++) {
    const float bval = bias[n0 + nt * 16 + nlan];
#pragma unroll
    for (int m = 0; m < 2; m++) {
#pragma unroll
      for (int i = 0; i < 8; i++) {
        float v = acc[m][nt][i] + bval;
        v = v > 0.f ? v : 0.f;
        const int row = m0 + m * 16 + i + hi * 8;
        const int col = n0 + nt * 16 + nlan;
        if (which != 2) {
          _Float16* outp = (which == 0) ? Qh : Kh;
          outp[(size_t)row * 512 + col] = (_Float16)v;
        } else {
          const int bb = row >> 12, s = row & 4095;  // batch, seq
          Vth[(size_t)bb * (512 * 4096) + (size_t)col * 4096 + s] =
              (_Float16)v;
        }
      }
    }
  }
}

// ---------------------------------------------------------------------------
// Stage 2: flash attention.  Block = (batch b, 32-row query strip), 8 waves.
// Per 128-key round:
//   - wave w computes a 32x16 score tile for keys [j0+16w, +16) over D=512:
//     per k-step 2 A-frags (Q rows 0-15 / 16-31 from LDS) share one K B-frag.
//   - online softmax across LDS (32 rows x 8 workers each).
//   - each wave accumulates a private 32x64 O-slice: per k-step 2 P A-frags
//     (LDS) share 4 V^T B-frags (global, contiguous) -> 8 WMMAs.
// grid = (128, 2); block = 256.
// ---------------------------------------------------------------------------
__global__ void __launch_bounds__(256) flash_attn(
    const _Float16* __restrict__ Qh, const _Float16* __restrict__ Kh,
    const _Float16* __restrict__ Vth, float* __restrict__ out) {
  constexpr int S = 4096;
  constexpr int D = 512;
  constexpr float SCALE = 0.044194173824159216f;  // 1/sqrt(512)

  __shared__ __align__(16) _Float16 Qs[32 * 512];  // 32 KB
  __shared__ __align__(16) float    Ss[32 * 128];  // 16 KB
  __shared__ __align__(16) _Float16 Ps[32 * 128];  // 8 KB
  __shared__ float red[32 * 8];
  __shared__ float mrow[32], lrow[32], alpha_s[32], mnew_s[32];

  const int tid  = threadIdx.x;
  const int wave = tid >> 5, lane = tid & 31;
  const int hi   = (lane >> 4) & 1, nlan = lane & 15;
  const int q0   = blockIdx.x * 32;
  const int b    = blockIdx.y;
  const int d0   = wave * 64;
  const int rrow = tid >> 3, rt = tid & 7;  // 32 rows x 8 workers

  // stage Q strip (32 x 512 f16) into LDS
  for (int i = tid * 8; i < 32 * 512; i += 256 * 8) {
    const int row = i >> 9, col = i & 511;
    *(v8h*)(&Qs[i]) =
        *(const v8h*)(Qh + ((size_t)(b * S + q0 + row) << 9) + col);
  }
  if (tid < 32) { mrow[tid] = -3.0e38f; lrow[tid] = 0.f; }
  __syncthreads();

  v8f zero = {};
  v8f acc[2][4];
#pragma unroll
  for (int m = 0; m < 2; m++)
#pragma unroll
    for (int nt = 0; nt < 4; nt++) acc[m][nt] = zero;

  for (int j0 = 0; j0 < S; j0 += 128) {
    // prefetch next round's K / V^T streams (global_prefetch_b8)
    if (j0 + 128 < S) {
      __builtin_prefetch(
          Kh + ((size_t)(b * S + j0 + 128 + wave * 16 + nlan) << 9) + hi * 256,
          0, 1);
      __builtin_prefetch(Vth + (size_t)b * (512 * 4096) +
                             (size_t)(d0 + nlan) * 4096 + j0 + 128 + hi * 64,
                         0, 1);
    }

    // ---- scores: 32 x 16 tile per wave, full-D contraction --------------
    v8f sc0 = zero, sc1 = zero;
    const _Float16* Kbase = Kh + ((size_t)(b * S + j0 + wave * 16) << 9);
#pragma unroll 4
    for (int k0 = 0; k0 < D; k0 += 32) {
      v16h a0 = load_a_frag(Qs, 512, k0, lane);
      v16h a1 = load_a_frag(Qs + (size_t)16 * 512, 512, k0, lane);
      v16h kb = load_b_frag(Kbase, 512, k0, lane);
      sc0 = WMMA_F32_F16(a0, kb, sc0);
      sc1 = WMMA_F32_F16(a1, kb, sc1);
    }
    const int colS = wave * 16 + nlan;
#pragma unroll
    for (int i = 0; i < 8; i++) {
      Ss[(i + hi * 8) * 128 + colS]      = sc0[i] * SCALE;
      Ss[(16 + i + hi * 8) * 128 + colS] = sc1[i] * SCALE;
    }
    __syncthreads();

    // ---- online softmax: row max -> m_new, alpha ------------------------
    float lmax = -3.0e38f;
#pragma unroll
    for (int u = 0; u < 16; u++)
      lmax = fmaxf(lmax, Ss[rrow * 128 + rt * 16 + u]);
    red[rrow * 8 + rt] = lmax;
    __syncthreads();
    if (tid < 32) {
      float tm = -3.0e38f;
      for (int u = 0; u < 8; u++) tm = fmaxf(tm, red[tid * 8 + u]);
      const float mo = mrow[tid];
      const float mn = fmaxf(mo, tm);
      mnew_s[tid]  = mn;
      alpha_s[tid] = __expf(mo - mn);
      mrow[tid]    = mn;
    }
    __syncthreads();

    // ---- P = exp(S - m_new) (f16 to LDS) + row sums ---------------------
    const float mn = mnew_s[rrow];
    float ls = 0.f;
#pragma unroll
    for (int u = 0; u < 16; u++) {
      const float p = __expf(Ss[rrow * 128 + rt * 16 + u] - mn);
      Ps[rrow * 128 + rt * 16 + u] = (_Float16)p;
      ls += p;
    }
    red[rrow * 8 + rt] = ls;
    __syncthreads();
    if (tid < 32) {
      float s = 0.f;
      for (int u = 0; u < 8; u++) s += red[tid * 8 + u];
      lrow[tid] = lrow[tid] * alpha_s[tid] + s;
    }
    __syncthreads();

    // ---- rescale O by alpha --------------------------------------------
    float av0[8], av1[8];
#pragma unroll
    for (int i = 0; i < 8; i++) {
      av0[i] = alpha_s[i + hi * 8];
      av1[i] = alpha_s[16 + i + hi * 8];
    }
#pragma unroll
    for (int nt = 0; nt < 4; nt++)
#pragma unroll
      for (int i = 0; i < 8; i++) {
        acc[0][nt][i] *= av0[i];
        acc[1][nt][i] *= av1[i];
      }

    // ---- O += P(32x128) @ V^T slabs (keys j0..j0+127, cols d0..d0+63) ---
    const _Float16* Vb =
        Vth + (size_t)b * (512 * 4096) + (size_t)d0 * 4096 + j0;
#pragma unroll
    for (int kk = 0; kk < 128; kk += 32) {
      v16h a0 = load_a_frag(Ps, 128, kk, lane);
      v16h a1 = load_a_frag(Ps + (size_t)16 * 128, 128, kk, lane);
#pragma unroll
      for (int nt = 0; nt < 4; nt++) {
        v16h vb = load_b_frag(Vb + (size_t)nt * 16 * 4096, 4096, kk, lane);
        acc[0][nt] = WMMA_F32_F16(a0, vb, acc[0][nt]);
        acc[1][nt] = WMMA_F32_F16(a1, vb, acc[1][nt]);
      }
    }
    __syncthreads();
  }

  // ---- finalize: O / l, store fp32 --------------------------------------
  float li0[8], li1[8];
#pragma unroll
  for (int i = 0; i < 8; i++) {
    li0[i] = 1.f / lrow[i + hi * 8];
    li1[i] = 1.f / lrow[16 + i + hi * 8];
  }
#pragma unroll
  for (int m = 0; m < 2; m++) {
#pragma unroll
    for (int i = 0; i < 8; i++) {
      const size_t rbase =
          ((size_t)(b * S + q0 + m * 16 + i + hi * 8)) << 9;
      const float sc = (m == 0) ? li0[i] : li1[i];
#pragma unroll
      for (int nt = 0; nt < 4; nt++) {
        out[rbase + d0 + nt * 16 + nlan] = acc[m][nt][i] * sc;
      }
    }
  }
}

// ---------------------------------------------------------------------------
extern "C" void kernel_launch(void* const* d_in, const int* in_sizes, int n_in,
                              void* d_out, int out_size, void* d_ws,
                              size_t ws_size, hipStream_t stream) {
  (void)in_sizes; (void)n_in; (void)out_size; (void)ws_size;

  const float* x1 = (const float*)d_in[0];
  const float* Wq = (const float*)d_in[1];
  const float* bq = (const float*)d_in[2];
  const float* Wk = (const float*)d_in[3];
  const float* bk = (const float*)d_in[4];
  const float* Wv = (const float*)d_in[5];
  const float* bv = (const float*)d_in[6];
  float* out = (float*)d_out;

  constexpr size_t SZ_X = (size_t)8192 * 512;  // elements
  constexpr size_t SZ_W = (size_t)512 * 512;

  char* ws = (char*)d_ws;
  _Float16* xh  = (_Float16*)(ws);
  _Float16* Wqh = (_Float16*)(ws + 2 * SZ_X);
  _Float16* Wkh = (_Float16*)(ws + 2 * (SZ_X + 1 * SZ_W));
  _Float16* Wvh = (_Float16*)(ws + 2 * (SZ_X + 2 * SZ_W));
  _Float16* Qh  = (_Float16*)(ws + 2 * (1 * SZ_X + 3 * SZ_W));
  _Float16* Kh  = (_Float16*)(ws + 2 * (2 * SZ_X + 3 * SZ_W));
  _Float16* Vth = (_Float16*)(ws + 2 * (3 * SZ_X + 3 * SZ_W));
  // total workspace used: ~33.5 MB

  cvt_f32_f16<<<2048, 256, 0, stream>>>(x1, xh, (int)SZ_X);
  cvt_f32_f16<<<256, 256, 0, stream>>>(Wq, Wqh, (int)SZ_W);
  cvt_f32_f16<<<256, 256, 0, stream>>>(Wk, Wkh, (int)SZ_W);
  cvt_f32_f16<<<256, 256, 0, stream>>>(Wv, Wvh, (int)SZ_W);

  qkv_gemm<<<768, 256, 0, stream>>>(xh, Wqh, Wkh, Wvh, bq, bk, bv, Qh, Kh,
                                    Vth);

  flash_attn<<<dim3(128, 2), 256, 0, stream>>>(Qh, Kh, Vth, out);
}
